// BasisGenNNTypeMulti_78082505441965
// MI455X (gfx1250) — matrix operand order repaired and checked by reading, not compiled, not run on hardware
//
#include <hip/hip_runtime.h>
#include <hip/hip_bf16.h>

// ---------------- problem constants ----------------
#define KAPN 16
#define MROWS 50000      // = 16 * 3125, no tail
#define MT16  3125       // 16-row tiles
#define DIM   128
#define H1DIM 384
#define H2DIM 256
#define NWAVES 2         // waves per block in MLP kernel

typedef __attribute__((ext_vector_type(16))) __bf16 v16bf;
typedef __attribute__((ext_vector_type(8)))  float  v8f;

union BFV { uint4 q[2]; v16bf v; };

// ---------------- workspace layout (bytes) ----------------
// xbf   : 50000*128  bf16  = 12,800,000
// w1t   : 16*128*384 bf16  =  1,572,864   (Wt[k][i][o], in-dim major)
// w2t   : 16*384*256 bf16  =  3,145,728
// w3t   : 16*256*128 bf16  =  1,048,576
// Phi   : 50000*16   f32   =  3,200,000
// part  : 256*256    f32   =    262,144
// Linv  : 256        f32   =      1,024
static constexpr size_t OFF_XBF  = 0;
static constexpr size_t OFF_W1T  = OFF_XBF + (size_t)MROWS * DIM * 2;
static constexpr size_t OFF_W2T  = OFF_W1T + (size_t)KAPN * DIM * H1DIM * 2;
static constexpr size_t OFF_W3T  = OFF_W2T + (size_t)KAPN * H1DIM * H2DIM * 2;
static constexpr size_t OFF_PHI  = OFF_W3T + (size_t)KAPN * H2DIM * DIM * 2;
static constexpr size_t OFF_PART = OFF_PHI + (size_t)MROWS * KAPN * 4;
static constexpr size_t OFF_LINV = OFF_PART + (size_t)256 * 256 * 4;

// ---------------- fragment loaders (CDNA5 WMMA bf16 layouts) ----------------
// A (16x32 bf16): lanes 0-15 rows M=0..15 hold K[kb+0..7] (v0-3) and K[kb+16..23] (v4-7);
//                 lanes 16-31 same rows hold K[kb+8..15] and K[kb+24..31].
__device__ __forceinline__ v16bf load_a_frag(const __bf16* p, int ld, int lane, int kb) {
    int row = lane & 15;
    int kh  = (lane >> 4) << 3;           // 0 or 8
    const __bf16* r0 = p + row * ld + kb + kh;
    BFV a;
    a.q[0] = *(const uint4*)(r0);
    a.q[1] = *(const uint4*)(r0 + 16);
    return a.v;
}

// B (32x16 bf16): lane L holds column N = nb + (L&15); K order matches A (kb+8h+0..7, kb+16+8h+0..7).
// Weights stored in-dim major (Wt[i][o], ld = DOUT); elements gathered with stride ld.
__device__ __forceinline__ v16bf load_b_frag(const __bf16* p, int ld, int lane, int kb, int nb) {
    const int n  = nb + (lane & 15);
    const int kh = (lane >> 4) << 3;
    v16bf b;
#pragma unroll
    for (int s = 0; s < 16; ++s)
        b[s] = p[(kb + kh + (s < 8 ? s : s + 8)) * ld + n];
    return b;
}

// ---------------- one MLP layer on a 16-row tile (one wave) ----------------
// D layout: VGPR r <-> row (mo+r), lane&15 <-> col; lanes>=16 are rows 8..15.
template<int DIN, int DOUT, bool LEAKY>
__device__ __forceinline__ void mlp_layer(const __bf16* __restrict__ in,
                                          const __bf16* __restrict__ wt,
                                          const float*  __restrict__ bias,
                                          __bf16* __restrict__ outb,   // row-major 16 x DOUT (ReLU path)
                                          float*  __restrict__ outf,   // col-major DOUT x 16 (Leaky path)
                                          int lane) {
    constexpr int NK = DIN / 32;
    v16bf a[NK];
#pragma unroll
    for (int kk = 0; kk < NK; ++kk)
        a[kk] = load_a_frag(in, DIN, lane, kk * 32);

    const int n_lane = lane & 15;
    const int mo     = (lane >> 4) << 3;  // 0 or 8

#pragma unroll
    for (int nb = 0; nb < DOUT; nb += 16) {
        v8f acc = {};
#pragma unroll
        for (int kk = 0; kk < NK; ++kk) {
            v16bf b = load_b_frag(wt, DOUT, lane, kk * 32, nb);
            acc = __builtin_amdgcn_wmma_f32_16x16x32_bf16(
                false, a[kk], false, b, (short)0, acc, false, false);
        }
        const float bv = bias[nb + n_lane];
#pragma unroll
        for (int r = 0; r < 8; ++r) {
            float v = acc[r] + bv;
            if constexpr (LEAKY) {
                v = (v > 0.f) ? v : 0.01f * v;
                outf[(nb + n_lane) * 16 + (mo + r)] = v;         // col-major f32
            } else {
                v = (v > 0.f) ? v : 0.f;
                outb[(mo + r) * DOUT + (nb + n_lane)] = (__bf16)v; // row-major bf16
            }
        }
    }
}

// ---------------- kernels ----------------
__global__ void cvt_x_bf16(const float* __restrict__ in, __bf16* __restrict__ out, int n) {
    int i = blockIdx.x * blockDim.x + threadIdx.x;
    if (i < n) out[i] = (__bf16)in[i];
}

// W[k][o][i] (f32) -> Wt[k][i][o] (bf16)
__global__ void cvt_w_transpose(const float* __restrict__ W, __bf16* __restrict__ Wt,
                                int dout, int din) {
    int per = dout * din;
    int i = blockIdx.x * blockDim.x + threadIdx.x;
    if (i >= KAPN * per) return;
    int k = i / per, r = i - k * per;
    int o = r / din, ii = r - o * din;
    Wt[(size_t)k * per + ii * dout + o] = (__bf16)W[i];
}

__global__ __launch_bounds__(NWAVES * 32)
void mlp_wmma_kernel(const __bf16* __restrict__ xbf,
                     const __bf16* __restrict__ w1t, const __bf16* __restrict__ w2t,
                     const __bf16* __restrict__ w3t,
                     const float* __restrict__ b1, const float* __restrict__ b2,
                     const float* __restrict__ b3,
                     const float* __restrict__ W4, const float* __restrict__ b4,
                     float* __restrict__ Phi) {
    __shared__ __align__(16) __bf16 sh1[NWAVES][16 * H1DIM];
    __shared__ __align__(16) __bf16 sh2[NWAVES][16 * H2DIM];
    __shared__ __align__(16) float  sh3[NWAVES][16 * DIM];   // col-major DIM x 16

    const int wave = threadIdx.x >> 5;
    const int lane = threadIdx.x & 31;
    const int tile = blockIdx.x * NWAVES + wave;
    const int k    = blockIdx.y;
    if (tile >= MT16) return;   // whole-wave exit: EXEC stays all-ones for WMMA

    const __bf16* xin = xbf + (size_t)tile * 16 * DIM;

    mlp_layer<DIM,   H1DIM, false>(xin,      w1t + (size_t)k * DIM * H1DIM,
                                   b1 + k * H1DIM, sh1[wave], nullptr, lane);
    mlp_layer<H1DIM, H2DIM, false>(sh1[wave], w2t + (size_t)k * H1DIM * H2DIM,
                                   b2 + k * H2DIM, sh2[wave], nullptr, lane);
    mlp_layer<H2DIM, DIM,   true >(sh2[wave], w3t + (size_t)k * H2DIM * DIM,
                                   b3 + k * DIM, nullptr, sh3[wave], lane);

    // Layer 4: phi[row] = h3[row,:] . W4[k,:] + b4[k]; lanes 0..15, conflict-free reads.
    if (lane < 16) {
        const float* w4k = W4 + k * DIM;
        float acc = 0.f;
#pragma unroll 8
        for (int i = 0; i < DIM; ++i)
            acc = fmaf(sh3[wave][i * 16 + lane], w4k[i], acc);
        Phi[(size_t)(tile * 16 + lane) * KAPN + k] = acc + b4[k];
    }
}

// Partial Gram matrix G = Phi^T Phi, deterministic (no atomics). 256 blocks x 256 threads.
__global__ void gram_partial(const float* __restrict__ Phi, float* __restrict__ part) {
    __shared__ float P[16][16];
    const int t = threadIdx.x;
    const int i = t & 15, j = t >> 4;
    float acc = 0.f;
    for (int m0 = blockIdx.x * 16; m0 < MROWS; m0 += gridDim.x * 16) {
        P[j][i] = Phi[(size_t)(m0 + j) * KAPN + i];
        __syncthreads();
#pragma unroll
        for (int mm = 0; mm < 16; ++mm)
            acc += P[mm][i] * P[mm][j];
        __syncthreads();
    }
    part[(size_t)blockIdx.x * 256 + t] = acc;
}

// Reduce partials -> G, Cholesky G = L L^T, write Linv = L^{-1}. One block.
__global__ void gram_chol_inv(const float* __restrict__ part, int nb, float* __restrict__ Linv) {
    __shared__ float G[256];
    const int t = threadIdx.x;
    float s = 0.f;
    for (int b = 0; b < nb; ++b) s += part[(size_t)b * 256 + t];
    G[t] = s;
    __syncthreads();
    if (t == 0) {
        float L[256], Li[256];
        for (int j = 0; j < 16; ++j) {
            float d = G[j * 16 + j];
            for (int p = 0; p < j; ++p) d -= L[j * 16 + p] * L[j * 16 + p];
            float lj = sqrtf(d);
            L[j * 16 + j] = lj;
            for (int i = j + 1; i < 16; ++i) {
                float v = G[i * 16 + j];
                for (int p = 0; p < j; ++p) v -= L[i * 16 + p] * L[j * 16 + p];
                L[i * 16 + j] = v / lj;
            }
        }
        for (int j = 0; j < 16; ++j) {
            Li[j * 16 + j] = 1.f / L[j * 16 + j];
            for (int i = 0; i < j; ++i) Li[i * 16 + j] = 0.f;
            for (int i = j + 1; i < 16; ++i) {
                float s2 = 0.f;
                for (int p = j; p < i; ++p) s2 += L[i * 16 + p] * Li[p * 16 + j];
                Li[i * 16 + j] = -s2 / L[i * 16 + i];
            }
        }
        for (int q = 0; q < 256; ++q) Linv[q] = Li[q];
    }
}

// Q = Phi * L^{-T}  (== classical Gram-Schmidt result). Q[m,j] = sum_{i<=j} Phi[m,i]*Linv[j,i]
__global__ void apply_q(const float* __restrict__ Phi, const float* __restrict__ Linv,
                        float* __restrict__ out) {
    int m = blockIdx.x * blockDim.x + threadIdx.x;
    if (m >= MROWS) return;
    float p[16];
#pragma unroll
    for (int i = 0; i < 16; ++i) p[i] = Phi[(size_t)m * KAPN + i];
#pragma unroll
    for (int j = 0; j < 16; ++j) {
        float s = 0.f;
#pragma unroll
        for (int i = 0; i < 16; ++i)
            if (i <= j) s += p[i] * Linv[j * 16 + i];
        out[(size_t)m * KAPN + j] = s;
    }
}

// ---------------- launcher ----------------
extern "C" void kernel_launch(void* const* d_in, const int* in_sizes, int n_in,
                              void* d_out, int out_size, void* d_ws, size_t ws_size,
                              hipStream_t stream) {
    (void)in_sizes; (void)n_in; (void)out_size; (void)ws_size;
    const float* x  = (const float*)d_in[0];
    const float* W1 = (const float*)d_in[1];
    const float* b1 = (const float*)d_in[2];
    const float* W2 = (const float*)d_in[3];
    const float* b2 = (const float*)d_in[4];
    const float* W3 = (const float*)d_in[5];
    const float* b3 = (const float*)d_in[6];
    const float* W4 = (const float*)d_in[7];
    const float* b4 = (const float*)d_in[8];
    float* out = (float*)d_out;

    char* ws = (char*)d_ws;
    __bf16* xbf  = (__bf16*)(ws + OFF_XBF);
    __bf16* w1t  = (__bf16*)(ws + OFF_W1T);
    __bf16* w2t  = (__bf16*)(ws + OFF_W2T);
    __bf16* w3t  = (__bf16*)(ws + OFF_W3T);
    float*  Phi  = (float*) (ws + OFF_PHI);
    float*  part = (float*) (ws + OFF_PART);
    float*  Linv = (float*) (ws + OFF_LINV);

    // 1) precision conversion + weight transposes (bf16, in-dim major)
    cvt_x_bf16<<<(MROWS * DIM + 255) / 256, 256, 0, stream>>>(x, xbf, MROWS * DIM);
    cvt_w_transpose<<<(KAPN * H1DIM * DIM  + 255) / 256, 256, 0, stream>>>(W1, w1t, H1DIM, DIM);
    cvt_w_transpose<<<(KAPN * H2DIM * H1DIM + 255) / 256, 256, 0, stream>>>(W2, w2t, H2DIM, H1DIM);
    cvt_w_transpose<<<(KAPN * DIM * H2DIM  + 255) / 256, 256, 0, stream>>>(W3, w3t, DIM, H2DIM);

    // 2) fused 4-layer MLP via WMMA: grid = (m-tiles / waves, heads)
    dim3 grid((MT16 + NWAVES - 1) / NWAVES, KAPN);
    mlp_wmma_kernel<<<grid, NWAVES * 32, 0, stream>>>(xbf, w1t, w2t, w3t,
                                                      b1, b2, b3, W4, b4, Phi);

    // 3) Gram-Schmidt == Cholesky-QR (deterministic, atomic-free)
    gram_partial<<<256, 256, 0, stream>>>(Phi, part);
    gram_chol_inv<<<1, 256, 0, stream>>>(part, 256, Linv);
    apply_q<<<(MROWS + 255) / 256, 256, 0, stream>>>(Phi, Linv, out);
}
